// PositionAttentionModule_62423054680109
// MI455X (gfx1250) — matrix operand order repaired and checked by reading, not compiled, not run on hardware
//
#include <hip/hip_runtime.h>
#include <hip/hip_bf16.h>

typedef __attribute__((ext_vector_type(16))) _Float16 v16h;
typedef __attribute__((ext_vector_type(8)))  _Float16 v8h;
typedef __attribute__((ext_vector_type(8)))  float    v8f;

#define BB   4
#define CC   64
#define NN   4096      // 64*64 spatial positions
#define QT   64        // query/key tile per workgroup
#define PADH 8         // f16 padding to stagger LDS banks

static __device__ __forceinline__ v16h cat16(v8h lo, v8h hi) {
  v16h r;
#pragma unroll
  for (int i = 0; i < 8; ++i) { r[i] = lo[i]; r[i + 8] = hi[i]; }
  return r;
}

// ---------------------------------------------------------------------------
// Pass 1: QKV projection (1x1 convs) as WMMA GEMMs.
//   Qh[b][n][o], Kh[b][n][o] (row = pixel, contiguous channels)  f16
//   Vt[b][o][n]              (row = channel, contiguous pixels)  f16
// ---------------------------------------------------------------------------
__global__ __launch_bounds__(128)
void qkv_proj_kernel(const float* __restrict__ x,
                     const float* __restrict__ W1, const float* __restrict__ b1,
                     const float* __restrict__ W2, const float* __restrict__ b2,
                     const float* __restrict__ W3, const float* __restrict__ b3,
                     _Float16* __restrict__ Qh, _Float16* __restrict__ Kh,
                     _Float16* __restrict__ Vt)
{
  __shared__ _Float16 xT[QT][CC + PADH];      // [n][c]  (transposed x tile)
  __shared__ _Float16 Wh[3][CC][CC + PADH];   // [o][c]  f16 weights

  const int t     = threadIdx.x;      // 0..127 (4 waves, wave32)
  const int lane  = t & 31;
  const int wave  = t >> 5;
  const int half  = lane >> 4;        // 0|1
  const int l16   = lane & 15;
  const int b     = blockIdx.y;
  const int nbase = blockIdx.x * QT;

  // Stage x tile: coalesced fp32 reads, transposed f16 stores
#pragma unroll
  for (int i = 0; i < 32; ++i) {
    int lin = t + 128 * i;            // 0..4095
    int c = lin >> 6, n = lin & 63;
    xT[n][c] = (_Float16)x[((size_t)(b * CC + c)) * NN + nbase + n];
  }
  // Stage weights converted to f16
  const float* Ws[3] = {W1, W2, W3};
#pragma unroll
  for (int w = 0; w < 3; ++w) {
#pragma unroll
    for (int i = 0; i < 32; ++i) {
      int lin = t + 128 * i;
      int o = lin >> 6, c = lin & 63;
      Wh[w][o][c] = (_Float16)Ws[w][o * CC + c];
    }
  }
  __syncthreads();

  // A fragments of x^T (rows n = 16*wave + l16), K-dim = channels
  const int nrow = 16 * wave + l16;
  v16h a[2];
#pragma unroll
  for (int at = 0; at < 2; ++at) {
    int c0 = at * 32 + half * 8;
    v8h lo = *reinterpret_cast<const v8h*>(&xT[nrow][c0]);
    v8h hi = *reinterpret_cast<const v8h*>(&xT[nrow][c0 + 16]);
    a[at] = cat16(lo, hi);
  }

  const float* bias[3] = {b1, b2, b3};
#pragma unroll
  for (int w = 0; w < 3; ++w) {
#pragma unroll
    for (int ot = 0; ot < 4; ++ot) {
      v8f acc = {};
#pragma unroll
      for (int ct = 0; ct < 2; ++ct) {
        // B fragment: lane holds column o, c values c0..c0+7 and c0+16..c0+23
        int orow = ot * 16 + l16;
        int c0   = ct * 32 + half * 8;
        v8h lo = *reinterpret_cast<const v8h*>(&Wh[w][orow][c0]);
        v8h hi = *reinterpret_cast<const v8h*>(&Wh[w][orow][c0 + 16]);
        acc = __builtin_amdgcn_wmma_f32_16x16x32_f16(
                  false, a[ct], false, cat16(lo, hi), (short)0, acc, false, false);
      }
      int   ocol = ot * 16 + l16;
      float bv   = bias[w][ocol];
#pragma unroll
      for (int v = 0; v < 8; ++v) {
        int n = nbase + 16 * wave + v + 8 * half;   // C/D layout: VGPR v -> row v / v+8
        _Float16 hval = (_Float16)(acc[v] + bv);
        if (w == 0)      Qh[((size_t)b * NN + n) * CC + ocol] = hval;
        else if (w == 1) Kh[((size_t)b * NN + n) * CC + ocol] = hval;
        else             Vt[((size_t)b * CC + ocol) * NN + n] = hval;
      }
    }
  }
}

// ---------------------------------------------------------------------------
// Pass 2: fused flash attention + residual.  out = x + alpha * (V @ softmax^T)
// One block = 64 query rows of one batch; 4 waves x 16 rows each.
// ---------------------------------------------------------------------------
__global__ __launch_bounds__(128)
void attn_kernel(const float* __restrict__ x, const float* __restrict__ alpha_p,
                 const _Float16* __restrict__ Qh, const _Float16* __restrict__ Kh,
                 const _Float16* __restrict__ Vt, float* __restrict__ out)
{
  __shared__ _Float16 Plds[4][16][QT + PADH];  // per-wave P tile (row-major)

  const int t     = threadIdx.x;
  const int lane  = t & 31;
  const int wave  = t >> 5;
  const int half  = lane >> 4;
  const int l16   = lane & 15;
  const int b     = blockIdx.y;
  const int qbase = blockIdx.x * QT;
  const float alpha = alpha_p[0];

  // Q A-fragments for this wave's 16 rows (K-dim = channels, k=64 -> 2 tiles)
  const int nrow = qbase + 16 * wave + l16;
  v16h aq[2];
#pragma unroll
  for (int at = 0; at < 2; ++at) {
    const _Float16* p = Qh + ((size_t)b * NN + nrow) * CC + at * 32 + half * 8;
    v8h lo = *reinterpret_cast<const v8h*>(p);
    v8h hi = *reinterpret_cast<const v8h*>(p + 16);
    aq[at] = cat16(lo, hi);
  }

  v8f oacc[4];
#pragma unroll
  for (int ct = 0; ct < 4; ++ct) oacc[ct] = (v8f){};
  float mrow[8], lrow[8];
#pragma unroll
  for (int v = 0; v < 8; ++v) { mrow[v] = -__builtin_inff(); lrow[v] = 0.f; }

  for (int j = 0; j < NN / QT; ++j) {
    const int kb = j * QT;

    // Prefetch next key/value block into cache (global_prefetch_b8)
    if (j + 1 < NN / QT) {
      __builtin_prefetch(Kh + ((size_t)b * NN + kb + QT + (size_t)l16 * 4) * CC, 0, 1);
      __builtin_prefetch(Vt + ((size_t)b * CC + (size_t)lane * 2) * NN + kb + QT, 0, 1);
    }

    // ---- scores: S = Q . K^T   (4 key sub-tiles of 16) ----
    v8f s[4];
#pragma unroll
    for (int kt = 0; kt < 4; ++kt) {
      v8f acc = {};
#pragma unroll
      for (int at = 0; at < 2; ++at) {
        int key = kb + kt * 16 + l16;                       // B: lane = column (key)
        const _Float16* p = Kh + ((size_t)b * NN + key) * CC + at * 32 + half * 8;
        v8h lo = *reinterpret_cast<const v8h*>(p);
        v8h hi = *reinterpret_cast<const v8h*>(p + 16);
        acc = __builtin_amdgcn_wmma_f32_16x16x32_f16(
                  false, aq[at], false, cat16(lo, hi), (short)0, acc, false, false);
      }
      s[kt] = acc;
    }

    // ---- online softmax: row max (rows live per-VGPR, cols across 16 lanes) ----
    float scale[8];
#pragma unroll
    for (int v = 0; v < 8; ++v) {
      float m = fmaxf(fmaxf(s[0][v], s[1][v]), fmaxf(s[2][v], s[3][v]));
#pragma unroll
      for (int off = 1; off < 16; off <<= 1)
        m = fmaxf(m, __shfl_xor(m, off, 32));   // stays inside 16-lane half
      float mn = fmaxf(mrow[v], m);
      scale[v] = __expf(mrow[v] - mn);          // first iter: exp(-inf)=0
      mrow[v]  = mn;
    }

    // ---- exponentiate + row sum ----
#pragma unroll
    for (int v = 0; v < 8; ++v) {
#pragma unroll
      for (int kt = 0; kt < 4; ++kt)
        s[kt][v] = __expf(s[kt][v] - mrow[v]);
      float r = (s[0][v] + s[1][v]) + (s[2][v] + s[3][v]);
#pragma unroll
      for (int off = 1; off < 16; off <<= 1)
        r += __shfl_xor(r, off, 32);
      lrow[v] = lrow[v] * scale[v] + r;
    }

    // ---- rescale running output ----
#pragma unroll
    for (int ct = 0; ct < 4; ++ct)
#pragma unroll
      for (int v = 0; v < 8; ++v)
        oacc[ct][v] *= scale[v];

    // ---- relayout P: C/D layout -> A layout via per-wave LDS tile ----
    __syncthreads();   // previous iteration's A-reads done before overwrite
#pragma unroll
    for (int kt = 0; kt < 4; ++kt)
#pragma unroll
      for (int v = 0; v < 8; ++v)
        Plds[wave][v + 8 * half][kt * 16 + l16] = (_Float16)s[kt][v];
    __syncthreads();

    v16h ap[2];
#pragma unroll
    for (int at = 0; at < 2; ++at) {
      int c0 = at * 32 + half * 8;
      v8h lo = *reinterpret_cast<const v8h*>(&Plds[wave][l16][c0]);
      v8h hi = *reinterpret_cast<const v8h*>(&Plds[wave][l16][c0 + 16]);
      ap[at] = cat16(lo, hi);
    }

    // ---- O += P @ V  (V^T stored channel-major: contiguous keys per lane) ----
#pragma unroll
    for (int ct = 0; ct < 4; ++ct) {
#pragma unroll
      for (int at = 0; at < 2; ++at) {
        int c  = ct * 16 + l16;                 // B: lane = column (channel)
        int k0 = kb + at * 32 + half * 8;
        const _Float16* p = Vt + ((size_t)b * CC + c) * NN + k0;
        v8h lo = *reinterpret_cast<const v8h*>(p);
        v8h hi = *reinterpret_cast<const v8h*>(p + 16);
        oacc[ct] = __builtin_amdgcn_wmma_f32_16x16x32_f16(
                       false, ap[at], false, cat16(lo, hi), (short)0, oacc[ct], false, false);
      }
    }
  }

  // ---- epilogue: out[b][c][n] = x + alpha * O[n][c] / l[n] ----
  float invl[8];
#pragma unroll
  for (int v = 0; v < 8; ++v) invl[v] = 1.f / lrow[v];
#pragma unroll
  for (int ct = 0; ct < 4; ++ct) {
    int c = ct * 16 + l16;
#pragma unroll
    for (int v = 0; v < 8; ++v) {
      int n = qbase + 16 * wave + v + 8 * half;
      size_t idx = ((size_t)b * CC + c) * NN + n;
      out[idx] = x[idx] + alpha * oacc[ct][v] * invl[v];
    }
  }
}

// ---------------------------------------------------------------------------
extern "C" void kernel_launch(void* const* d_in, const int* in_sizes, int n_in,
                              void* d_out, int out_size, void* d_ws, size_t ws_size,
                              hipStream_t stream) {
  const float* x  = (const float*)d_in[0];
  const float* W1 = (const float*)d_in[1];
  const float* b1 = (const float*)d_in[2];
  const float* W2 = (const float*)d_in[3];
  const float* b2 = (const float*)d_in[4];
  const float* W3 = (const float*)d_in[5];
  const float* b3 = (const float*)d_in[6];
  const float* al = (const float*)d_in[7];
  float* out = (float*)d_out;

  _Float16* Qh = (_Float16*)d_ws;                       // 2 MiB
  _Float16* Kh = Qh + (size_t)BB * NN * CC;             // 2 MiB
  _Float16* Vt = Kh + (size_t)BB * NN * CC;             // 2 MiB

  dim3 grid(NN / QT, BB);   // 64 query tiles x 4 batches
  qkv_proj_kernel<<<grid, 128, 0, stream>>>(x, W1, b1, W2, b2, W3, b3, Qh, Kh, Vt);
  attn_kernel<<<grid, 128, 0, stream>>>(x, al, Qh, Kh, Vt, out);
}
